// KVCacheSamAttention_43473658970634
// MI455X (gfx1250) — matrix operand order repaired and checked by reading, not compiled, not run on hardware
//
#include <hip/hip_runtime.h>

typedef _Float16 h8   __attribute__((ext_vector_type(8)));
typedef _Float16 v16h __attribute__((ext_vector_type(16)));
typedef float    v8f  __attribute__((ext_vector_type(8)));
typedef unsigned int u32x4 __attribute__((ext_vector_type(4)));
typedef int          i32x4 __attribute__((ext_vector_type(4)));
typedef int          i32x8 __attribute__((ext_vector_type(8)));

#define B_   2
#define H_   16
#define SQ_  2048
#define SKV_ 2048
#define D_   64
#define BLK_KV 64
#define WAVES_PER_WG 8
#define ROWS_PER_WAVE 16
#define ROWS_PER_WG (WAVES_PER_WG * ROWS_PER_WAVE)   // 128
#define LOG2E_F 1.4426950408889634f

#if __has_builtin(__builtin_amdgcn_tensor_load_to_lds) && \
    __has_builtin(__builtin_amdgcn_s_wait_tensorcnt)
#define HAVE_TDM 1
#else
#define HAVE_TDM 0
#endif

// D = A(16x32 f16) x B(32x16 f16) + C(16x16 f32)
static __device__ __forceinline__ v8f wmma_f16(v16h a, v16h b, v8f c) {
    return __builtin_amdgcn_wmma_f32_16x16x32_f16(false, a, false, b, (short)0, c,
                                                  false, false);
}

static __device__ __forceinline__ v16h cat8(h8 lo, h8 hi) {
    return __builtin_shufflevector(lo, hi, 0, 1, 2, 3, 4, 5, 6, 7,
                                           8, 9, 10, 11, 12, 13, 14, 15);
}

// Butterfly max across 16 lanes with DPP (no LDS traffic). Masks {1,2,7,15}
// (quad_perm xor1, quad_perm xor2, row_half_mirror, row_mirror) span the group.
static __device__ __forceinline__ float fmax_x16(float x) {
    float t;
    t = __int_as_float(__builtin_amdgcn_update_dpp(
            __float_as_int(x), __float_as_int(x), 0x0B1, 0xF, 0xF, false));
    x = fmaxf(x, t);
    t = __int_as_float(__builtin_amdgcn_update_dpp(
            __float_as_int(x), __float_as_int(x), 0x04E, 0xF, 0xF, false));
    x = fmaxf(x, t);
    t = __int_as_float(__builtin_amdgcn_update_dpp(
            __float_as_int(x), __float_as_int(x), 0x141, 0xF, 0xF, false));
    x = fmaxf(x, t);
    t = __int_as_float(__builtin_amdgcn_update_dpp(
            __float_as_int(x), __float_as_int(x), 0x140, 0xF, 0xF, false));
    x = fmaxf(x, t);
    return x;
}

// Low 32 bits of a generic pointer into the LDS aperture = LDS byte offset.
static __device__ __forceinline__ unsigned lds_off(const void* p) {
    return (unsigned)(unsigned long long)(uintptr_t)p;
}

// ---------------------------------------------------------------------------
// Tensor Data Mover: async 2-D f16 tile load (global -> LDS). D# per
// CDNA5 ISA. dims/strides in elements (data_size=2B).
// ---------------------------------------------------------------------------
static __device__ __forceinline__ void tdm_load_2d(const void* gaddr,
                                                   unsigned lds_addr,
                                                   unsigned tensor_d0,
                                                   unsigned tensor_d1,
                                                   unsigned tile_d0,
                                                   unsigned tile_d1,
                                                   unsigned stride0) {
#if HAVE_TDM
    const unsigned long long ga = (unsigned long long)(uintptr_t)gaddr;
    u32x4 g0;
    g0[0] = 1u;                                       // count=1, user mode
    g0[1] = lds_addr;                                 // lds_addr (bytes)
    g0[2] = (unsigned)(ga & 0xFFFFFFFFu);             // global_addr[31:0]
    g0[3] = (unsigned)((ga >> 32) & 0x01FFFFFFu) |    // global_addr[56:32]
            (2u << 30);                               // type = 2 ("image")
    i32x8 g1;
    g1[0] = (int)(1u << 16);                          // data_size=1 -> 2 bytes
    g1[1] = (int)((tensor_d0 & 0xFFFFu) << 16);       // tensor_dim0[15:0]@48
    g1[2] = (int)(((tensor_d0 >> 16) & 0xFFFFu) |
                  ((tensor_d1 & 0xFFFFu) << 16));     // dim0[31:16], dim1[15:0]
    g1[3] = (int)(((tensor_d1 >> 16) & 0xFFFFu) |
                  ((tile_d0 & 0xFFFFu) << 16));       // dim1[31:16], tile0
    g1[4] = (int)(tile_d1 & 0xFFFFu);                 // tile1; tile2=0 (unused)
    g1[5] = (int)stride0;                             // tensor_dim0_stride[31:0]
    g1[6] = 0;
    g1[7] = 0;
    const i32x4 z4 = (i32x4)0;
    const i32x8 z8 = (i32x8)0;
    __builtin_amdgcn_tensor_load_to_lds(g0, g1, z4, z4, z8, 0);
#endif
}

static __device__ __forceinline__ void tdm_wait0() {
#if HAVE_TDM
    __builtin_amdgcn_s_wait_tensorcnt(0);
#endif
}

// ---------------------------------------------------------------------------
// Prep kernels (run once): K f32 -> f16; V f32 -> f16 transposed [bh][d][kv];
// mask f32 -> f32 * log2(e) (base-2 softmax domain).
// ---------------------------------------------------------------------------
__global__ __launch_bounds__(256)
void preconvert(const float* __restrict__ k, const float* __restrict__ v,
                _Float16* __restrict__ kh, _Float16* __restrict__ vth)
{
    __shared__ __align__(16) _Float16 vt[64][72];   // +8 pad halfs per row

    const int tid = threadIdx.x;
    const int bh  = blockIdx.x / (SKV_ / 64);
    const int kv0 = (blockIdx.x % (SKV_ / 64)) * 64;

    const float* kb  = k  + ((size_t)bh * SKV_ + kv0) * D_;
    const float* vb  = v  + ((size_t)bh * SKV_ + kv0) * D_;
    _Float16*    khb = kh + ((size_t)bh * SKV_ + kv0) * D_;
    _Float16*    vtb = vth + (size_t)bh * D_ * SKV_ + kv0;

    #pragma unroll
    for (int g = 0; g < 2; ++g) {
        const int base = (tid + g * 256) * 8;
        float4 a = *(const float4*)(kb + base);
        float4 b = *(const float4*)(kb + base + 4);
        h8 o;
        o[0] = (_Float16)a.x; o[1] = (_Float16)a.y;
        o[2] = (_Float16)a.z; o[3] = (_Float16)a.w;
        o[4] = (_Float16)b.x; o[5] = (_Float16)b.y;
        o[6] = (_Float16)b.z; o[7] = (_Float16)b.w;
        *(h8*)(khb + base) = o;
    }
    #pragma unroll
    for (int g = 0; g < 4; ++g) {
        const int idx  = tid + g * 256;
        const int row  = idx >> 4;
        const int col4 = (idx & 15) << 2;
        float4 x = *(const float4*)(vb + (size_t)row * D_ + col4);
        vt[col4 + 0][row] = (_Float16)x.x;
        vt[col4 + 1][row] = (_Float16)x.y;
        vt[col4 + 2][row] = (_Float16)x.z;
        vt[col4 + 3][row] = (_Float16)x.w;
    }
    __syncthreads();
    #pragma unroll
    for (int g = 0; g < 2; ++g) {
        const int chunk = tid + g * 256;
        const int d  = chunk >> 3;
        const int c8 = (chunk & 7) * 8;
        *(h8*)(vtb + (size_t)d * SKV_ + c8) = *(const h8*)&vt[d][c8];
    }
}

__global__ __launch_bounds__(256)
void mask_scale(const float* __restrict__ mask, float* __restrict__ msc)
{
    const size_t base = ((size_t)blockIdx.x * 256 + threadIdx.x) * 8;
    float4 a = *(const float4*)(mask + base);
    float4 b = *(const float4*)(mask + base + 4);
    a.x *= LOG2E_F; a.y *= LOG2E_F; a.z *= LOG2E_F; a.w *= LOG2E_F;
    b.x *= LOG2E_F; b.y *= LOG2E_F; b.z *= LOG2E_F; b.w *= LOG2E_F;
    *(float4*)(msc + base)     = a;
    *(float4*)(msc + base + 4) = b;
}

// ---------------------------------------------------------------------------
// Fused flash attention (base-2 softmax domain). PRE=true: pre-converted f16
// K/Vt staged by TDM async tensor loads (b128 reg copies if no TDM builtin),
// mask prescaled by log2e. PRE=false: everything computed in-loop.
// ---------------------------------------------------------------------------
template <bool PRE>
__global__ __launch_bounds__(256)
void fa_fwd(const float* __restrict__ q, const float* __restrict__ k,
            const float* __restrict__ v, const float* __restrict__ mask,
            const int* __restrict__ scale_p, float* __restrict__ out,
            const _Float16* __restrict__ kh, const _Float16* __restrict__ vth)
{
    // Kt: [buf][kv][d].  Vt: [buf][d][kv]; rows 64..79 = ones tile so P x ones
    // accumulates the softmax row-sum as a 5th output tile.
    __shared__ __align__(16) _Float16 Kt[2][BLK_KV][D_];            // 16KB
    __shared__ __align__(16) _Float16 Vt[2][D_ + 16][BLK_KV];       // 20KB
    __shared__ __align__(16) _Float16 Pst[WAVES_PER_WG][16][BLK_KV];// 16KB

    const int tid  = threadIdx.x;
    const int wave = tid >> 5;
    const int lane = tid & 31;
    const int n    = lane & 15;
    const int hl   = lane >> 4;

    const int qtiles = SQ_ / ROWS_PER_WG;
    const int bh    = blockIdx.x / qtiles;
    const int qtile = blockIdx.x % qtiles;
    const int q0    = qtile * ROWS_PER_WG + wave * ROWS_PER_WAVE;

    const float inv_scale = 1.0f / (float)scale_p[0];
    const float c1 = inv_scale * LOG2E_F;       // QK scale folded with log2e

    const float* qbase  = q   + (size_t)bh * SQ_  * D_;
    float*       obase  = out + (size_t)bh * SQ_  * D_;
    const _Float16* khb = PRE ? kh  + (size_t)bh * SKV_ * D_ : nullptr;
    const _Float16* vtb = PRE ? vth + (size_t)bh * D_ * SKV_ : nullptr;
    const float* kbase  = k + (size_t)bh * SKV_ * D_;
    const float* vbase  = v + (size_t)bh * SKV_ * D_;

    // Ones tile (both buffers), written once, never overwritten.
    #pragma unroll
    for (int g = 0; g < 8; ++g) {
        const int idx = tid + g * 256;
        const int b   = idx >> 10;
        const int r   = (idx >> 6) & 15;
        const int c   = idx & 63;
        Vt[b][D_ + r][c] = (_Float16)1.0f;
    }

    // ---- Q tile -> two 16x32 f16 A-fragments ------------------------------
    v16h qa[2];
    {
        const float* qr = qbase + (size_t)(q0 + n) * D_;
        #pragma unroll
        for (int f = 0; f < 2; ++f) {
            #pragma unroll
            for (int c = 0; c < 2; ++c) {
                const int d0 = 32 * f + 16 * hl + 8 * c;
                float4 x = *(const float4*)(qr + d0);
                float4 y = *(const float4*)(qr + d0 + 4);
                qa[f][8 * c + 0] = (_Float16)x.x;
                qa[f][8 * c + 1] = (_Float16)x.y;
                qa[f][8 * c + 2] = (_Float16)x.z;
                qa[f][8 * c + 3] = (_Float16)x.w;
                qa[f][8 * c + 4] = (_Float16)y.x;
                qa[f][8 * c + 5] = (_Float16)y.y;
                qa[f][8 * c + 6] = (_Float16)y.z;
                qa[f][8 * c + 7] = (_Float16)y.w;
            }
        }
    }

    v8f  o_acc[5];                       // [4] = softmax denominator tile
    float m_i[8];                        // running max, base-2 domain
    #pragma unroll
    for (int t = 0; t < 5; ++t) o_acc[t] = (v8f)0.0f;
    #pragma unroll
    for (int i = 0; i < 8; ++i) m_i[i] = -__builtin_inff();

    uint4 kreg[2], vreg[2];

    auto load_tile = [&](int kv0) {      // no-TDM PRE path: global f16 -> regs
        #pragma unroll
        for (int g = 0; g < 2; ++g)
            kreg[g] = *(const uint4*)((const char*)(khb + (size_t)kv0 * D_) +
                                      (tid + g * 256) * 16);
        #pragma unroll
        for (int g = 0; g < 2; ++g) {
            const int chunk = tid + g * 256;
            const int d  = chunk >> 3;
            const int c8 = (chunk & 7) * 8;
            vreg[g] = *(const uint4*)(vtb + (size_t)d * SKV_ + kv0 + c8);
        }
    };
    auto store_tile = [&](int buf) {
        #pragma unroll
        for (int g = 0; g < 2; ++g)
            *(uint4*)((char*)&Kt[buf][0][0] + (tid + g * 256) * 16) = kreg[g];
        #pragma unroll
        for (int g = 0; g < 2; ++g) {
            const int chunk = tid + g * 256;
            const int d  = chunk >> 3;
            const int c8 = (chunk & 7) * 8;
            *(uint4*)&Vt[buf][d][c8] = vreg[g];
        }
    };
    auto conv_tile = [&](int kv0) {      // PRE=false fallback
        #pragma unroll
        for (int it = 0; it < 4; ++it) {
            const int idx  = tid + it * 256;
            const int row  = idx >> 4;
            const int col4 = (idx & 15) << 2;
            float4 kx = *(const float4*)(kbase + (size_t)(kv0 + row) * D_ + col4);
            Kt[0][row][col4 + 0] = (_Float16)kx.x;
            Kt[0][row][col4 + 1] = (_Float16)kx.y;
            Kt[0][row][col4 + 2] = (_Float16)kx.z;
            Kt[0][row][col4 + 3] = (_Float16)kx.w;
            float4 vx = *(const float4*)(vbase + (size_t)(kv0 + row) * D_ + col4);
            Vt[0][col4 + 0][row] = (_Float16)vx.x;
            Vt[0][col4 + 1][row] = (_Float16)vx.y;
            Vt[0][col4 + 2][row] = (_Float16)vx.z;
            Vt[0][col4 + 3][row] = (_Float16)vx.w;
        }
    };
    auto issue_tiles = [&](int kv0, int buf) {   // TDM path: wave 0 only
        tdm_load_2d(khb + (size_t)kv0 * D_, lds_off(&Kt[buf][0][0]),
                    D_, SKV_, D_, BLK_KV, D_);
        tdm_load_2d(vtb + kv0, lds_off(&Vt[buf][0][0]),
                    SKV_, D_, BLK_KV, D_, SKV_);
    };

    auto compute = [&](int buf, int kv0) {
        // ---- S = Q * K^T ---------------------------------------------------
        v8f s[4];
        #pragma unroll
        for (int t = 0; t < 4; ++t) {
            const _Float16* kr = &Kt[buf][16 * t + n][0];
            v16h kb0 = cat8(*(const h8*)(kr + 16 * hl),
                            *(const h8*)(kr + 16 * hl + 8));
            v16h kb1 = cat8(*(const h8*)(kr + 32 + 16 * hl),
                            *(const h8*)(kr + 32 + 16 * hl + 8));
            v8f acc = (v8f)0.0f;
            acc  = wmma_f16(qa[0], kb0, acc);
            s[t] = wmma_f16(qa[1], kb1, acc);
        }
        // ---- scale + additive mask, base-2 domain --------------------------
        #pragma unroll
        for (int t = 0; t < 4; ++t) {
            v8f mv;
            #pragma unroll
            for (int i = 0; i < 8; ++i)
                mv[i] = mask[(size_t)(q0 + i + 8 * hl) * SKV_ + kv0 + 16 * t + n];
            if constexpr (PRE) {
                s[t] = s[t] * c1 + mv;                    // mask prescaled
            } else {
                s[t] = (s[t] * inv_scale + mv) * LOG2E_F; // fallback fixup
            }
        }
        // ---- online-softmax max (DPP butterfly) ----------------------------
        v8f av;
        #pragma unroll
        for (int i = 0; i < 8; ++i) {
            float mx = fmaxf(fmaxf(s[0][i], s[1][i]), fmaxf(s[2][i], s[3][i]));
            mx = fmax_x16(mx);
            const float mnew = fmaxf(m_i[i], mx);
            av[i]  = __builtin_exp2f(m_i[i] - mnew);      // bare v_exp_f32
            m_i[i] = mnew;
        }
        // ---- P = exp2(S - m) staged to LDS for C->A layout flip ------------
        #pragma unroll
        for (int t = 0; t < 4; ++t) {
            #pragma unroll
            for (int i = 0; i < 8; ++i) {
                const float p = __builtin_exp2f(s[t][i] - m_i[i]);
                Pst[wave][i + 8 * hl][16 * t + n] = (_Float16)p;
            }
        }
        // ---- rescale accumulators (vector multiply) ------------------------
        #pragma unroll
        for (int t = 0; t < 5; ++t) o_acc[t] = o_acc[t] * av;
        // ---- reload P as A-fragments (per-wave region) ---------------------
        v16h pa[2];
        {
            const _Float16* pr = &Pst[wave][n][0];
            #pragma unroll
            for (int f = 0; f < 2; ++f)
                pa[f] = cat8(*(const h8*)(pr + 32 * f + 16 * hl),
                             *(const h8*)(pr + 32 * f + 16 * hl + 8));
        }
        // ---- O += P*V; tile 4 = ones tile -> row sums ----------------------
        #pragma unroll
        for (int t = 0; t < 5; ++t) {
            const _Float16* vr = &Vt[buf][16 * t + n][0];
            v16h vb0 = cat8(*(const h8*)(vr + 16 * hl),
                            *(const h8*)(vr + 16 * hl + 8));
            v16h vb1 = cat8(*(const h8*)(vr + 32 + 16 * hl),
                            *(const h8*)(vr + 32 + 16 * hl + 8));
            o_acc[t] = wmma_f16(pa[0], vb0, o_acc[t]);
            o_acc[t] = wmma_f16(pa[1], vb1, o_acc[t]);
        }
    };

    if constexpr (PRE) {
        const int iters = SKV_ / BLK_KV;
        if constexpr (HAVE_TDM) {
            if (wave == 0) {
                issue_tiles(0, 0);
                tdm_wait0();
            }
            __syncthreads();
            for (int it = 0; it < iters; ++it) {
                const int kv0 = it * BLK_KV;
                const int cur = it & 1;
                const bool more = (it + 1) < iters;
                if (more && wave == 0) issue_tiles(kv0 + BLK_KV, cur ^ 1);
                compute(cur, kv0);
                if (more && wave == 0) tdm_wait0();   // DMA done before barrier
                __syncthreads();
            }
        } else {
            load_tile(0);
            store_tile(0);
            __syncthreads();
            for (int it = 0; it < iters; ++it) {
                const int kv0 = it * BLK_KV;
                const int cur = it & 1;
                const bool more = (it + 1) < iters;
                if (more) load_tile(kv0 + BLK_KV);
                compute(cur, kv0);
                if (more) store_tile(cur ^ 1);
                __syncthreads();
            }
        }
    } else {
        for (int kv0 = 0; kv0 < SKV_; kv0 += BLK_KV) {
            conv_tile(kv0);
            __syncthreads();
            compute(0, kv0);
            __syncthreads();
        }
    }

    // ---- normalize (denominator from the ones-tile accumulator) -----------
    #pragma unroll
    for (int i = 0; i < 8; ++i) {
        const float invl = 1.0f / o_acc[4][i];
        float* orow = obase + (size_t)(q0 + i + 8 * hl) * D_;
        #pragma unroll
        for (int t = 0; t < 4; ++t)
            orow[16 * t + n] = o_acc[t][i] * invl;
    }
}

extern "C" void kernel_launch(void* const* d_in, const int* in_sizes, int n_in,
                              void* d_out, int out_size, void* d_ws, size_t ws_size,
                              hipStream_t stream) {
    const float* q     = (const float*)d_in[0];
    const float* k     = (const float*)d_in[1];
    const float* v     = (const float*)d_in[2];
    const float* mask  = (const float*)d_in[3];
    const int*   scale = (const int*)d_in[4];
    float*       out   = (float*)d_out;

    dim3 grid(B_ * H_ * (SQ_ / ROWS_PER_WG));   // 512 workgroups
    dim3 block(256);                            // 8 wave32s

    const size_t elems   = (size_t)B_ * H_ * SKV_ * D_;
    const size_t mselems = (size_t)SQ_ * SKV_;
    const size_t needed  = 2 * elems * sizeof(_Float16) + mselems * sizeof(float);

    if (ws_size >= needed) {
        _Float16* kh  = (_Float16*)d_ws;
        _Float16* vth = kh + elems;
        float*    msc = (float*)(vth + elems);
        preconvert<<<dim3(B_ * H_ * (SKV_ / 64)), block, 0, stream>>>(k, v, kh, vth);
        mask_scale<<<dim3((unsigned)(mselems / (256 * 8))), block, 0, stream>>>(mask, msc);
        fa_fwd<true><<<grid, block, 0, stream>>>(q, k, v, msc, scale, out, kh, vth);
    } else {
        fa_fwd<false><<<grid, block, 0, stream>>>(q, k, v, mask, scale, out,
                                                  nullptr, nullptr);
    }
}
